// KANConv2d_2121713844780
// MI455X (gfx1250) — matrix operand order repaired and checked by reading, not compiled, not run on hardware
//
#include <hip/hip_runtime.h>
#include <hip/hip_bf16.h>

// ---------------------------------------------------------------------------
// KANConv2d on gfx1250 (MI455X), fused as one bf16 WMMA GEMM:
//   K = 576 features * 9 (silu + 8 cubic B-spline bases) = 5184
//   A (16384 x 5184) activations, built on the fly in LDS (closed-form
//       uniform cubic B-spline weights, 4-nonzero scatter)
//   B (5184 x 128)  combined weights, pre-packed into WMMA B-fragment order
//   D (16384 x 128) -> reshaped (4,128,64,64) fp32 output
// Chunk K layout (must match pack kernel): k = ck*288 + c*32 + il,
//   feature i = ck*32 + il, c = 0 -> silu/base path, c = 1..8 -> spline bases.
// ---------------------------------------------------------------------------

typedef __attribute__((ext_vector_type(16))) __bf16 v16bf;
typedef __attribute__((ext_vector_type(8)))  float  v8f;

#define IN_C     64
#define OUT_C    128
#define HW       64
#define IN_FEAT  576          // 64 * 9 taps
#define KTOT     5184         // 576 * 9
#define KTILES   162          // KTOT / 32
#define NTILES   8            // 128 / 16
#define CHUNK_K  288          // 9 coeffs * 32 features per chunk
#define NCHUNKS  18           // 576 / 32
#define PACK_ELEMS (KTILES * NTILES * 32 * 16)   // 663552 bf16
#define ADUMMY   (16 * CHUNK_K)                  // dummy scatter sink in LDS

// float -> bf16 round-to-nearest-even (bit pattern), always compiles
__device__ __forceinline__ unsigned short f2bf(float f) {
  unsigned int u = __float_as_uint(f);
  u += 0x7FFFu + ((u >> 16) & 1u);
  return (unsigned short)(u >> 16);
}

// ---------------------------------------------------------------------------
// Pack combined weight matrix into per-lane WMMA B-fragment order:
// block (ktile, ntile) -> 32 lanes x 16 bf16 contiguous.
// B 32x16 bf16 layout: lanes 0-15 hold K=0..7+16..23 (2 per VGPR), lanes 16-31
// hold K=8..15+24..31; N = lane % 16.
// ---------------------------------------------------------------------------
__global__ __launch_bounds__(256) void pack_weights(
    const float* __restrict__ bw,   // (128, 576)
    const float* __restrict__ sw,   // (128, 576, 8)
    const float* __restrict__ sc,   // (128, 576)
    unsigned short* __restrict__ wp) {
  int idx = blockIdx.x * 256 + threadIdx.x;
  if (idx >= PACK_ELEMS) return;
  int j    = idx & 15;          // element within lane's 16-bf16 chunk
  int lane = (idx >> 4) & 31;
  int wv   = (idx >> 9) & 7;    // n-tile
  int ktg  = idx >> 12;         // global k-tile 0..161
  int klocal = ((j & 8) << 1) + ((lane >> 4) << 3) + (j & 7);   // 0..31
  int k  = ktg * 32 + klocal;            // 0..5183
  int n  = lane & 15;
  int oc = wv * 16 + n;
  // planar chunk layout: k = ck*288 + c*32 + il
  int ck = k / CHUNK_K;
  int r  = k - ck * CHUNK_K;
  int c  = r >> 5;              // 0..8
  int il = r & 31;
  int i  = ck * 32 + il;        // unfold feature 0..575
  float v;
  if (c == 0) v = bw[oc * IN_FEAT + i];
  else        v = sw[(oc * IN_FEAT + i) * 8 + (c - 1)] * sc[oc * IN_FEAT + i];
  wp[idx] = f2bf(v);
}

// ---------------------------------------------------------------------------
// Main fused kernel: one block = 16 pixels (contiguous in one image row),
// 8 waves; wave wv owns output-channel tile [wv*16, wv*16+16).
// ---------------------------------------------------------------------------
__global__ __launch_bounds__(256) void kan_conv_wmma(
    const float* __restrict__ xin,          // (4,64,64,64)
    const unsigned short* __restrict__ wp,  // packed bf16 B fragments
    float* __restrict__ out) {              // (4,128,64,64)
  __shared__ float xs[IN_C * 3 * 18];                            // halo tile
  __shared__ __align__(16) unsigned short As[16 * CHUNK_K + 32]; // A panel + sink

  const int tid  = threadIdx.x;
  const int P0   = blockIdx.x * 16;   // first pixel of this block
  const int bimg = P0 >> 12;          // image index
  const int rem  = P0 & 4095;         // h*64 + w0 within image
  const int h    = rem >> 6;
  const int w0   = rem & 63;          // multiple of 16

  // Stage 64ch x 3rows x 18cols halo (zero-padded) into LDS.
  for (int t = tid; t < IN_C * 54; t += 256) {
    int cc = t / 54;
    int r  = t - cc * 54;
    int dh = r / 18;
    int dw = r - dh * 18;
    int hh = h - 1 + dh;
    int ww = w0 - 1 + dw;
    float v = 0.0f;
    if (hh >= 0 && hh < HW && ww >= 0 && ww < HW)
      v = xin[((bimg * IN_C + cc) * HW + hh) * HW + ww];
    xs[t] = v;
  }

  const int lane = tid & 31;
  const int wv   = tid >> 5;   // wave id = n-tile
  v8f acc = {};

  for (int ck = 0; ck < NCHUNKS; ++ck) {
    // ---- zero A panel (576 x ds_store_b128 across the block) ----
    uint4 z4 = {0u, 0u, 0u, 0u};
    for (int t = tid; t < (16 * CHUNK_K * 2) / 16; t += 256)
      ((uint4*)As)[t] = z4;
    __syncthreads();

    // ---- activation phase: 16 pixels x 32 features, 2 tasks per thread ----
#pragma unroll
    for (int tt = 0; tt < 2; ++tt) {
      int q   = tid + tt * 256;       // 0..511
      int il  = q & 31;               // feature in chunk (lane-contiguous)
      int pl  = (q >> 5) & 15;        // pixel in block
      int i   = ck * 32 + il;         // global unfold feature
      int cin = i / 9;
      int tap = i - cin * 9;
      int kh  = tap / 3;
      int kw  = tap - kh * 3;
      float x = xs[cin * 54 + kh * 18 + pl + kw];

      // SiLU -> c = 0 slot
      float s = x * (1.0f / (1.0f + __expf(-x)));
      As[pl * CHUNK_K + il] = f2bf(s);

      // Closed-form uniform cubic B-spline: knots T[j] = 0.4*j - 2.2,
      // t = (x - T[0]) / h, span j = floor(t), local u = t - j.
      // Nonzero bases: index i = j-3..j (clipped to [0,8)) with weights:
      //   w0 = u^3/6 (i=j), w1, w2, w3 = (1-u)^3/6 (i=j-3); sum = 1.
      float t  = __fmaf_rn(x, 2.5f, 5.5f);
      float jf = floorf(t);
      float u  = t - jf;
      int   j  = (int)jf;
      float u2 = u * u;
      float u3 = u2 * u;
      float om = 1.0f - u;
      float p0 = u3 * 0.16666667f;
      float p3 = om * om * om * 0.16666667f;
      float p2 = 0.66666667f - u2 + 0.5f * u3;
      float p1 = 1.0f - p0 - p2 - p3;
      float w[4] = {p0, p1, p2, p3};    // w[r] -> basis index j - r
#pragma unroll
      for (int r = 0; r < 4; ++r) {
        int bi = j - r;                               // basis index 0..7 if valid
        bool valid = ((unsigned)bi) < 8u;             // also false for |x|>=2.2
        int addr = valid ? (pl * CHUNK_K + (bi + 1) * 32 + il)
                         : (ADUMMY + lane);           // branch-free sink
        As[addr] = f2bf(w[r]);
      }
    }
    __syncthreads();

    // ---- WMMA phase: 9 k-tiles of 32 over this chunk ----
    // A 16x32 bf16 fragment: lanes 0-15 -> M=lane, K[0..7]+K[16..23];
    //                        lanes 16-31 -> M=lane-16, K[8..15]+K[24..31].
    const int m   = lane & 15;
    const int sel = lane >> 4;
    const unsigned short* arow  = &As[m * CHUNK_K + sel * 8];
    const unsigned short* bbase = wp + (((ck * 9) * 8 + wv) * 32 + lane) * 16;
#pragma unroll
    for (int kt = 0; kt < 9; ++kt) {
      union { v16bf v; uint4 q[2]; } Af, Bf;
      const unsigned short* ap = arow + kt * 32;
      Af.q[0] = *(const uint4*)(ap);        // ds_load_b128
      Af.q[1] = *(const uint4*)(ap + 16);   // ds_load_b128
      const unsigned short* bp = bbase + kt * (NTILES * 512);
      Bf.q[0] = ((const uint4*)bp)[0];      // global_load_b128
      Bf.q[1] = ((const uint4*)bp)[1];      // global_load_b128
      if (ck + 1 < NCHUNKS)                 // keep next chunk's B L2/L0-hot
        __builtin_prefetch(bp + 9 * NTILES * 512, 0, 1);  // global_prefetch_b8
      acc = __builtin_amdgcn_wmma_f32_16x16x32_bf16(
          false, Af.v, false, Bf.v, (short)0, acc, false, false);
    }
    __syncthreads();   // protect As before next chunk re-zeroes it
  }

  // ---- store D: C/D layout -> lane = N (out channel), VGPR r = M (pixel) ----
  const int n  = lane & 15;
  const int mb = (lane >> 4) << 3;       // lanes 16-31 hold M = r + 8
  const int oc = wv * 16 + n;
  float* op = out + ((bimg * OUT_C + oc) << 12) + rem;
#pragma unroll
  for (int r = 0; r < 8; ++r) op[mb + r] = acc[r];
}

extern "C" void kernel_launch(void* const* d_in, const int* in_sizes, int n_in,
                              void* d_out, int out_size, void* d_ws, size_t ws_size,
                              hipStream_t stream) {
  const float* x  = (const float*)d_in[0];   // (4,64,64,64)
  const float* bw = (const float*)d_in[1];   // (128,576)
  const float* sw = (const float*)d_in[2];   // (128,576,8)
  const float* sc = (const float*)d_in[3];   // (128,576)
  unsigned short* wpk = (unsigned short*)d_ws;  // 1.33 MB packed bf16 weights
  float* out = (float*)d_out;                   // (4,128,64,64)

  pack_weights<<<PACK_ELEMS / 256, 256, 0, stream>>>(bw, sw, sc, wpk);
  kan_conv_wmma<<<(4 * HW * HW) / 16, 256, 0, stream>>>(x, wpk, out);
}
